// ATTDecodeBlock_80427557585265
// MI455X (gfx1250) — matrix operand order repaired and checked by reading, not compiled, not run on hardware
//
#include <hip/hip_runtime.h>
#include <math.h>

#define B_ 2
#define S_ 8192
#define H_ 32
#define HD_ 128
#define D_ 4096
#define FF_ 11008
#define NCHUNK 8
#define CH (S_ / NCHUNK) /* 1024 */

typedef __attribute__((ext_vector_type(16))) __bf16 v16bf;
typedef __attribute__((ext_vector_type(8)))  float  v8f;

__device__ __forceinline__ float wave_reduce_sum(float v) {
#pragma unroll
  for (int off = 16; off > 0; off >>= 1) v += __shfl_xor(v, off, 32);
  return v;
}

// ---------------- RMSNorm ----------------
__global__ void rmsnorm_kernel(const float* __restrict__ x, const float* __restrict__ w,
                               float* __restrict__ out) {
  int b = blockIdx.x;
  const float* xb = x + (size_t)b * D_;
  __shared__ float red[8];
  float ss = 0.f;
  for (int i = threadIdx.x; i < D_; i += blockDim.x) { float v = xb[i]; ss += v * v; }
  ss = wave_reduce_sum(ss);
  int wid = threadIdx.x >> 5, lane = threadIdx.x & 31;
  if (lane == 0) red[wid] = ss;
  __syncthreads();
  float tot = 0.f;
#pragma unroll
  for (int i = 0; i < 8; ++i) tot += red[i];
  float inv = rsqrtf(tot / (float)D_ + 1e-6f);
  for (int i = threadIdx.x; i < D_; i += blockDim.x)
    out[(size_t)b * D_ + i] = xb[i] * inv * w[i];
}

// ---------------- QKV GEMV: out(b,3D) = Wqkv(3D,D) @ xn(b,D) ----------------
__global__ void gemv_qkv(const float* __restrict__ W, const float* __restrict__ xn,
                         float* __restrict__ out) {
  int lane = threadIdx.x & 31;
  int gw = blockIdx.x * (blockDim.x >> 5) + (threadIdx.x >> 5);
  int nw = gridDim.x * (blockDim.x >> 5);
  const float4* x0 = (const float4*)(xn);
  const float4* x1 = (const float4*)(xn + D_);
  for (int r = gw; r < 3 * D_; r += nw) {
    const float4* wr = (const float4*)(W + (size_t)r * D_);
    if (r + nw < 3 * D_)
      __builtin_prefetch((const char*)(W + (size_t)(r + nw) * D_) + lane * 512, 0, 0);
    float a0 = 0.f, a1 = 0.f;
    for (int c = lane; c < D_ / 4; c += 32) {
      float4 wv = wr[c];
      float4 xv = x0[c];
      a0 += wv.x * xv.x + wv.y * xv.y + wv.z * xv.z + wv.w * xv.w;
      float4 yv = x1[c];
      a1 += wv.x * yv.x + wv.y * yv.y + wv.z * yv.z + wv.w * yv.w;
    }
    a0 = wave_reduce_sum(a0);
    a1 = wave_reduce_sum(a1);
    if (lane == 0) { out[r] = a0; out[3 * D_ + r] = a1; }
  }
}

// ---------------- RoPE + write K/V row into output caches ----------------
__global__ void rope_kv(const float* __restrict__ qkv, const float* __restrict__ cosT,
                        const float* __restrict__ sinT, const int* __restrict__ posp,
                        float* __restrict__ qout, float* __restrict__ kout,
                        float* __restrict__ vout) {
  int pos = posp[0];
  int bh = blockIdx.x;
  int b = bh / H_, h = bh % H_;
  int d = threadIdx.x; // 128 threads
  size_t base = (size_t)b * 3 * D_ + (size_t)h * HD_;
  float q = qkv[base + d];
  float k = qkv[base + D_ + d];
  float v = qkv[base + 2 * D_ + d];
  int dm = d & 63;
  float c = cosT[(size_t)pos * 64 + dm];
  float s = sinT[(size_t)pos * 64 + dm];
  float qr = (d < 64) ? -qkv[base + d + 64] : qkv[base + d - 64];
  float kr = (d < 64) ? -qkv[base + D_ + d + 64] : qkv[base + D_ + d - 64];
  qout[(size_t)b * D_ + (size_t)h * HD_ + d] = q * c + qr * s;
  size_t off = (((size_t)b * S_ + pos) * H_ + h) * HD_ + d;
  kout[off] = k * c + kr * s;
  vout[off] = v;
}

// ---------------- Flash-decode partial: per (b,h,chunk) ----------------
__global__ void attn_partial(const float* __restrict__ q, const float* __restrict__ K,
                             const float* __restrict__ V, const int* __restrict__ posp,
                             float* __restrict__ pm, float* __restrict__ pl,
                             float* __restrict__ pacc) {
  int pos = posp[0];
  int blk = blockIdx.x;
  int chunk = blk % NCHUNK;
  int h = (blk / NCHUNK) % H_;
  int b = blk / (NCHUNK * H_);
  int tid = threadIdx.x, lane = tid & 31, wid = tid >> 5;

  __shared__ float qsh[HD_];
  __shared__ float m_s[8], l_s[8];
  __shared__ float acc_s[8][HD_];

  if (tid < HD_) qsh[tid] = q[((size_t)b * H_ + h) * HD_ + tid];
  __syncthreads();

  const float scale = 0.08838834764831845f; // 1/sqrt(128)
  int m16 = lane & 15, hi = lane >> 4;

  // B operand: q in column 0 (lane0 holds K=0..7,16..23 of each 32-chunk, lane16 holds K=8..15,24..31)
  v16bf bq[4];
#pragma unroll
  for (int kc = 0; kc < 4; ++kc) {
#pragma unroll
    for (int j = 0; j < 16; ++j) {
      float qv = (m16 == 0) ? qsh[kc * 32 + 8 * hi + (j & 7) + ((j >> 3) << 4)] : 0.0f;
      bq[kc][j] = (__bf16)qv;
    }
  }

  float m_run = -1e30f, l_run = 0.f;
  float acc0 = 0.f, acc1 = 0.f, acc2 = 0.f, acc3 = 0.f;
  int s_begin = chunk * CH;
  int s_last = (pos < s_begin + CH - 1) ? pos : (s_begin + CH - 1); // inclusive
  const size_t rowstride = (size_t)H_ * HD_;
  const float* Kb = K + ((size_t)b * S_) * rowstride + (size_t)h * HD_;
  const float* Vb = V + ((size_t)b * S_) * rowstride + (size_t)h * HD_;

  for (int s0 = s_begin + wid * 16; s0 <= s_last; s0 += 8 * 16) {
    // A operand: 16 K-rows (positions s0..s0+15), fp32 -> bf16, documented 16-bit A layout
    v8f c = {0.f, 0.f, 0.f, 0.f, 0.f, 0.f, 0.f, 0.f};
#pragma unroll
    for (int kc = 0; kc < 4; ++kc) {
      const float* krow = Kb + (size_t)(s0 + m16) * rowstride + kc * 32;
      v16bf a;
#pragma unroll
      for (int j = 0; j < 8; ++j) {
        a[j]     = (__bf16)krow[8 * hi + j];
        a[8 + j] = (__bf16)krow[16 + 8 * hi + j];
      }
      c = __builtin_amdgcn_wmma_f32_16x16x32_bf16(false, a, false, bq[kc],
                                                  (short)0, c, false, false);
    }
    // scores live in D column 0: lane0 (M=0..7), lane16 (M=8..15)
    float sc[16];
#pragma unroll
    for (int j = 0; j < 8; ++j) {
      sc[j]     = __shfl(c[j], 0, 32) * scale;
      sc[8 + j] = __shfl(c[j], 16, 32) * scale;
    }
    int nvalid = s_last - s0 + 1;
    if (nvalid > 16) nvalid = 16;
    float tmax = -1e30f;
#pragma unroll
    for (int j = 0; j < 16; ++j)
      if (j < nvalid) tmax = fmaxf(tmax, sc[j]);
    float newm = fmaxf(m_run, tmax);
    float corr = __expf(m_run - newm);
    acc0 *= corr; acc1 *= corr; acc2 *= corr; acc3 *= corr; l_run *= corr;
#pragma unroll
    for (int j = 0; j < 16; ++j) {
      if (j < nvalid) {
        float p = __expf(sc[j] - newm);
        l_run += p;
        const float4 vv = *(const float4*)(Vb + (size_t)(s0 + j) * rowstride + 4 * lane);
        acc0 += p * vv.x; acc1 += p * vv.y; acc2 += p * vv.z; acc3 += p * vv.w;
      }
    }
    m_run = newm;
  }

  if (lane == 0) { m_s[wid] = m_run; l_s[wid] = l_run; }
  acc_s[wid][4 * lane + 0] = acc0;
  acc_s[wid][4 * lane + 1] = acc1;
  acc_s[wid][4 * lane + 2] = acc2;
  acc_s[wid][4 * lane + 3] = acc3;
  __syncthreads();

  if (tid < HD_) {
    float M = -1e30f;
#pragma unroll
    for (int w = 0; w < 8; ++w) M = fmaxf(M, m_s[w]);
    float L = 0.f, Av = 0.f;
#pragma unroll
    for (int w = 0; w < 8; ++w) {
      float f = __expf(m_s[w] - M);
      L += l_s[w] * f;
      Av += acc_s[w][tid] * f;
    }
    int pidx = (b * H_ + h) * NCHUNK + chunk;
    pacc[(size_t)pidx * HD_ + tid] = Av;
    if (tid == 0) { pm[pidx] = M; pl[pidx] = L; }
  }
}

// ---------------- Combine chunk partials -> ctx(b, D) ----------------
__global__ void attn_reduce(const float* __restrict__ pm, const float* __restrict__ pl,
                            const float* __restrict__ pacc, float* __restrict__ ctx) {
  int bh = blockIdx.x;
  int tid = threadIdx.x; // 128
  int base = bh * NCHUNK;
  float M = -1e30f;
#pragma unroll
  for (int i = 0; i < NCHUNK; ++i) M = fmaxf(M, pm[base + i]);
  float L = 0.f, Av = 0.f;
#pragma unroll
  for (int i = 0; i < NCHUNK; ++i) {
    float f = __expf(pm[base + i] - M);
    L += pl[base + i] * f;
    Av += pacc[(size_t)(base + i) * HD_ + tid] * f;
  }
  ctx[(size_t)bh * HD_ + tid] = Av / L;
}

// ---------------- out_proj GEMV + residual ----------------
__global__ void gemv_oproj_res(const float* __restrict__ W, const float* __restrict__ ctx,
                               const float* __restrict__ x, float* __restrict__ x1) {
  int lane = threadIdx.x & 31;
  int gw = blockIdx.x * (blockDim.x >> 5) + (threadIdx.x >> 5);
  int nw = gridDim.x * (blockDim.x >> 5);
  const float4* c0 = (const float4*)(ctx);
  const float4* c1 = (const float4*)(ctx + D_);
  for (int r = gw; r < D_; r += nw) {
    const float4* wr = (const float4*)(W + (size_t)r * D_);
    if (r + nw < D_)
      __builtin_prefetch((const char*)(W + (size_t)(r + nw) * D_) + lane * 512, 0, 0);
    float a0 = 0.f, a1 = 0.f;
    for (int c = lane; c < D_ / 4; c += 32) {
      float4 wv = wr[c];
      float4 xv = c0[c];
      a0 += wv.x * xv.x + wv.y * xv.y + wv.z * xv.z + wv.w * xv.w;
      float4 yv = c1[c];
      a1 += wv.x * yv.x + wv.y * yv.y + wv.z * yv.z + wv.w * yv.w;
    }
    a0 = wave_reduce_sum(a0);
    a1 = wave_reduce_sum(a1);
    if (lane == 0) {
      x1[r] = x[r] + a0;
      x1[D_ + r] = x[D_ + r] + a1;
    }
  }
}

// ---------------- gate/up fused GEMV: ff = silu(h@l1^T) * (h@l2^T) ----------------
__global__ void gemv_gateup(const float* __restrict__ W1, const float* __restrict__ W2,
                            const float* __restrict__ h, float* __restrict__ ff) {
  int lane = threadIdx.x & 31;
  int gw = blockIdx.x * (blockDim.x >> 5) + (threadIdx.x >> 5);
  int nw = gridDim.x * (blockDim.x >> 5);
  const float4* h0 = (const float4*)(h);
  const float4* h1 = (const float4*)(h + D_);
  for (int r = gw; r < FF_; r += nw) {
    const float4* w1 = (const float4*)(W1 + (size_t)r * D_);
    const float4* w2 = (const float4*)(W2 + (size_t)r * D_);
    if (r + nw < FF_) {
      __builtin_prefetch((const char*)(W1 + (size_t)(r + nw) * D_) + lane * 512, 0, 0);
      __builtin_prefetch((const char*)(W2 + (size_t)(r + nw) * D_) + lane * 512, 0, 0);
    }
    float g0 = 0.f, g1 = 0.f, u0 = 0.f, u1 = 0.f;
    for (int c = lane; c < D_ / 4; c += 32) {
      float4 a = w1[c];
      float4 bq = w2[c];
      float4 xv = h0[c];
      float4 yv = h1[c];
      g0 += a.x * xv.x + a.y * xv.y + a.z * xv.z + a.w * xv.w;
      g1 += a.x * yv.x + a.y * yv.y + a.z * yv.z + a.w * yv.w;
      u0 += bq.x * xv.x + bq.y * xv.y + bq.z * xv.z + bq.w * xv.w;
      u1 += bq.x * yv.x + bq.y * yv.y + bq.z * yv.z + bq.w * yv.w;
    }
    g0 = wave_reduce_sum(g0);
    g1 = wave_reduce_sum(g1);
    u0 = wave_reduce_sum(u0);
    u1 = wave_reduce_sum(u1);
    if (lane == 0) {
      ff[r]       = (g0 / (1.f + __expf(-g0))) * u0;
      ff[FF_ + r] = (g1 / (1.f + __expf(-g1))) * u1;
    }
  }
}

// ---------------- down proj GEMV + residual -> final x ----------------
__global__ void gemv_down_res(const float* __restrict__ W3, const float* __restrict__ ff,
                              const float* __restrict__ x1, float* __restrict__ xout) {
  int lane = threadIdx.x & 31;
  int gw = blockIdx.x * (blockDim.x >> 5) + (threadIdx.x >> 5);
  int nw = gridDim.x * (blockDim.x >> 5);
  const float4* f0 = (const float4*)(ff);
  const float4* f1 = (const float4*)(ff + FF_);
  for (int r = gw; r < D_; r += nw) {
    const float4* wr = (const float4*)(W3 + (size_t)r * FF_);
    if (r + nw < D_)
      __builtin_prefetch((const char*)(W3 + (size_t)(r + nw) * FF_) + lane * 1024, 0, 0);
    float a0 = 0.f, a1 = 0.f;
    for (int c = lane; c < FF_ / 4; c += 32) {
      float4 wv = wr[c];
      float4 xv = f0[c];
      a0 += wv.x * xv.x + wv.y * xv.y + wv.z * xv.z + wv.w * xv.w;
      float4 yv = f1[c];
      a1 += wv.x * yv.x + wv.y * yv.y + wv.z * yv.z + wv.w * yv.w;
    }
    a0 = wave_reduce_sum(a0);
    a1 = wave_reduce_sum(a1);
    if (lane == 0) {
      xout[r] = x1[r] + a0;
      xout[D_ + r] = x1[D_ + r] + a1;
    }
  }
}

extern "C" void kernel_launch(void* const* d_in, const int* in_sizes, int n_in,
                              void* d_out, int out_size, void* d_ws, size_t ws_size,
                              hipStream_t stream) {
  (void)in_sizes; (void)n_in; (void)out_size; (void)ws_size;
  const float* x        = (const float*)d_in[0];
  const int*   posp     = (const int*)d_in[1];   // int64 on LE: low dword is the value
  const float* k_cache  = (const float*)d_in[2];
  const float* v_cache  = (const float*)d_in[3];
  const float* pre_w    = (const float*)d_in[4];
  const float* post_w   = (const float*)d_in[5];
  const float* Wqkv     = (const float*)d_in[6];
  const float* Wout     = (const float*)d_in[7];
  const float* l1       = (const float*)d_in[8];
  const float* l2       = (const float*)d_in[9];
  const float* l3       = (const float*)d_in[10];
  const float* rope_cos = (const float*)d_in[11];
  const float* rope_sin = (const float*)d_in[12];

  float* out_x = (float*)d_out;
  float* out_k = out_x + (size_t)B_ * D_;
  float* out_v = out_k + (size_t)B_ * S_ * H_ * HD_;
  const size_t cache_bytes = (size_t)B_ * S_ * H_ * HD_ * sizeof(float);

  // workspace layout (floats)
  float* w     = (float*)d_ws;
  float* xn    = w;                         // 2*D
  float* qkv   = xn + (size_t)B_ * D_;      // 2*3D
  float* qrope = qkv + (size_t)B_ * 3 * D_; // 2*D
  float* ctx   = qrope + (size_t)B_ * D_;   // 2*D
  float* x1    = ctx + (size_t)B_ * D_;     // 2*D
  float* hbuf  = x1 + (size_t)B_ * D_;      // 2*D
  float* ffb   = hbuf + (size_t)B_ * D_;    // 2*FF
  float* pm    = ffb + (size_t)B_ * FF_;    // B*H*NCHUNK
  float* pl    = pm + (size_t)B_ * H_ * NCHUNK;
  float* pacc  = pl + (size_t)B_ * H_ * NCHUNK; // B*H*NCHUNK*HD

  // 1) out-of-place cache copies (updated row written on top by rope_kv)
  hipMemcpyAsync(out_k, k_cache, cache_bytes, hipMemcpyDeviceToDevice, stream);
  hipMemcpyAsync(out_v, v_cache, cache_bytes, hipMemcpyDeviceToDevice, stream);

  // 2) pre-norm
  rmsnorm_kernel<<<B_, 256, 0, stream>>>(x, pre_w, xn);
  // 3) QKV projection (weight row read once for both batches)
  gemv_qkv<<<512, 256, 0, stream>>>(Wqkv, xn, qkv);
  // 4) RoPE + cache row update (into output caches)
  rope_kv<<<B_ * H_, 128, 0, stream>>>(qkv, rope_cos, rope_sin, posp, qrope, out_k, out_v);
  // 5) flash-decode attention: per (b,h,chunk) partials (WMMA QK^T), then combine
  attn_partial<<<B_ * H_ * NCHUNK, 256, 0, stream>>>(qrope, out_k, out_v, posp, pm, pl, pacc);
  attn_reduce<<<B_ * H_, HD_, 0, stream>>>(pm, pl, pacc, ctx);
  // 6) out_proj + residual
  gemv_oproj_res<<<256, 256, 0, stream>>>(Wout, ctx, x, x1);
  // 7) post-norm
  rmsnorm_kernel<<<B_, 256, 0, stream>>>(x1, post_w, hbuf);
  // 8) MLP gate/up fused
  gemv_gateup<<<512, 256, 0, stream>>>(l1, l2, hbuf, ffb);
  // 9) down proj + residual -> final x
  gemv_down_res<<<256, 256, 0, stream>>>(l3, ffb, x1, out_x);
}
